// EfficientAttention_74113955660238
// MI455X (gfx1250) — matrix operand order repaired and checked by reading, not compiled, not run on hardware
//
#include <hip/hip_runtime.h>
#include <hip/hip_bf16.h>

// ---------------------------------------------------------------------------
// Types
// ---------------------------------------------------------------------------
typedef __bf16 bf16_t;
typedef __attribute__((ext_vector_type(16))) __bf16 v16bf;
typedef __attribute__((ext_vector_type(8)))  __bf16 v8bf;
typedef __attribute__((ext_vector_type(8)))  float  v8f;

#define DEVINL __device__ __forceinline__

static constexpr int B_  = 2;
static constexpr int S_  = 2048;
static constexpr int E_  = 1024;
static constexpr int H_  = 16;
static constexpr int D_  = 64;
static constexpr int N_  = B_ * S_;   // 4096 rows

// ---------------------------------------------------------------------------
// bf16 helpers
// ---------------------------------------------------------------------------
DEVINL bf16_t f2bf(float f) {
  unsigned u = __float_as_uint(f);
  u += 0x7FFFu + ((u >> 16) & 1u);          // round-to-nearest-even
  unsigned short h = (unsigned short)(u >> 16);
  bf16_t b;
  __builtin_memcpy(&b, &h, sizeof(h));
  return b;
}

DEVINL v8f wmma_bf16(v16bf a, v16bf b, v8f c) {
  // D = A(16x32) * B(32x16) + C, fp32 accumulate
  return __builtin_amdgcn_wmma_f32_16x16x32_bf16(
      /*neg_a=*/false, a, /*neg_b=*/false, b,
      /*c_mod=*/(short)0, c, /*reuse_a=*/false, /*reuse_b=*/false);
}

// Async 16-byte global -> LDS copy (per-lane), tracked by ASYNCcnt.
// LDS destination address = low 32 bits of the flat shared-aperture address.
DEVINL void async_copy_b128(void* lds_dst, const void* gsrc) {
  unsigned ldsoff = (unsigned)(uintptr_t)lds_dst;
  asm volatile("global_load_async_to_lds_b128 %0, %1, off"
               :: "v"(ldsoff), "v"(gsrc) : "memory");
}
DEVINL void async_wait0() {
  asm volatile("s_wait_asynccnt 0" ::: "memory");
}

// A-matrix fragment (16x32 bf16), row-major source with row stride `ld` elems.
// ISA layout: lane<16 holds row m=lane, K = {0..7, 16..23}; lane>=16: K = {8..15, 24..31}.
DEVINL v16bf frag_a(const bf16_t* base, int ld) {
  const int lane = threadIdx.x & 31;
  const int r  = lane & 15;
  const int ko = (lane >> 4) << 3;          // 0 or 8
  v16bf f;
  ((v8bf*)&f)[0] = *(const v8bf*)(base + r * ld + ko);
  ((v8bf*)&f)[1] = *(const v8bf*)(base + r * ld + 16 + ko);
  return f;
}

// B-matrix fragment (32x16 bf16) from an N-major tile: base[n][k], stride `ld`.
// ISA layout: lane<16 holds col n=lane, K = {0..7, 16..23}; lane>=16: K = {8..15, 24..31}.
DEVINL v16bf frag_b(const bf16_t* base, int ld) {
  const int lane = threadIdx.x & 31;
  const int r  = lane & 15;
  const int ko = (lane >> 4) << 3;          // 0 or 8
  v16bf f;
  ((v8bf*)&f)[0] = *(const v8bf*)(base + r * ld + ko);
  ((v8bf*)&f)[1] = *(const v8bf*)(base + r * ld + 16 + ko);
  return f;
}

DEVINL void store_out(float* p, float v)  { *p = v; }
DEVINL void store_out(bf16_t* p, float v) { *p = f2bf(v); }

// ---------------------------------------------------------------------------
// fp32 -> bf16 conversion
// ---------------------------------------------------------------------------
__global__ void cvt_f32_bf16(const float* __restrict__ in, bf16_t* __restrict__ out, int n) {
  int i = blockIdx.x * blockDim.x + threadIdx.x;
  if (i < n) out[i] = f2bf(in[i]);
}

// ---------------------------------------------------------------------------
// GEMM: C[M,N] = A[M,K] * Bw[K,N]   (A,Bw bf16 row-major; C bf16 or f32)
// 256 threads (8 waves), 128x128 block tile, wave tile 32x64, K-step 32.
// Per wave per K-step: 2 A-frags x 4 B-frags -> 8 WMMA.
// A-tile staged with GLOBAL_LOAD_ASYNC_TO_LDS_B128; B-tile transposed via regs.
// ---------------------------------------------------------------------------
template <typename OutT>
__global__ __launch_bounds__(256)
void gemm_bf16_kernel(const bf16_t* __restrict__ A, const bf16_t* __restrict__ Bw,
                      OutT* __restrict__ C, int M, int Nn, int K) {
  __shared__ __attribute__((aligned(16))) bf16_t As[128][40];  // row-major tile
  __shared__ __attribute__((aligned(16))) bf16_t Bs[128][40];  // N-major (transposed) tile
  const int tid  = threadIdx.x;
  const int lane = tid & 31;
  const int w    = tid >> 5;                 // 8 waves: 4 row-groups x 2 col-groups
  const int m0 = blockIdx.y * 128, n0 = blockIdx.x * 128;
  const int wm = (w >> 1) * 32,   wn = (w & 1) * 64;

  v8f acc[2][4] = {};
  for (int kk = 0; kk < K; kk += 32) {
    // stage A 128x32 (row-major): async global->LDS, 16B per lane, 2 per thread
    #pragma unroll
    for (int i = tid; i < 128 * 4; i += 256) {
      int m = i >> 2, kc = (i & 3) << 3;
      async_copy_b128(&As[m][kc], &A[(size_t)(m0 + m) * K + kk + kc]);
      if (kk + 32 < K)
        __builtin_prefetch(&A[(size_t)(m0 + m) * K + kk + 32 + kc], 0, 1);
    }
    // stage B 32x128, transposed into Bs[n][k]
    for (int i = tid; i < 32 * 16; i += 256) {
      int k = i >> 4, nc = (i & 15) << 3;
      v8bf t = *(const v8bf*)&Bw[(size_t)(kk + k) * Nn + n0 + nc];
      #pragma unroll
      for (int j = 0; j < 8; ++j) Bs[nc + j][k] = t[j];
    }
    async_wait0();
    __syncthreads();

    // load all fragments first, then issue the WMMA chain (overlap ds with wmma)
    v16bf a0 = frag_a(&As[wm][0],      40);
    v16bf a1 = frag_a(&As[wm + 16][0], 40);
    v16bf bf[4];
    #pragma unroll
    for (int j = 0; j < 4; ++j) bf[j] = frag_b(&Bs[wn + 16 * j][0], 40);
    #pragma unroll
    for (int j = 0; j < 4; ++j) {
      acc[0][j] = wmma_bf16(a0, bf[j], acc[0][j]);
      acc[1][j] = wmma_bf16(a1, bf[j], acc[1][j]);
    }
    __syncthreads();
  }

  // C/D layout: n = lane&15 ; row m = r + 8*(lane>=16) for VGPR r
  const int half = lane >> 4, n = lane & 15;
  #pragma unroll
  for (int i = 0; i < 2; ++i)
    #pragma unroll
    for (int j = 0; j < 4; ++j)
      #pragma unroll
      for (int r = 0; r < 8; ++r) {
        int m  = m0 + wm + i * 16 + half * 8 + r;
        int nn = n0 + wn + j * 16 + n;
        store_out(&C[(size_t)m * Nn + nn], acc[i][j][r]);
      }
}

// ---------------------------------------------------------------------------
// Flash attention (causal). Grid: (S/128, H, B), 256 threads (8 waves).
// Block handles 128 query rows of one (b,h); key tiles of 64.
// Per wave per key tile: 8 score WMMA + 8 PV WMMA.
// K-tile staged async; V-tile transposed via regs.
// ---------------------------------------------------------------------------
__global__ __launch_bounds__(256)
void attn_kernel(const bf16_t* __restrict__ Q, const bf16_t* __restrict__ Kt,
                 const bf16_t* __restrict__ V, bf16_t* __restrict__ ctx) {
  __shared__ __attribute__((aligned(16))) bf16_t Ks[64][72];      // [key][d]   (B-frag for QK^T)
  __shared__ __attribute__((aligned(16))) bf16_t Vs[64][72];      // [d][key]   (B-frag for PV)
  __shared__ __attribute__((aligned(16))) bf16_t Ps[8][16][72];   // per-wave P tile (16x64)

  const int tid = threadIdx.x, lane = tid & 31, w = tid >> 5;
  const int bq = blockIdx.z, h = blockIdx.y, q0 = blockIdx.x * 128;
  const size_t rowbase = (size_t)bq * S_ * E_;
  const int col0 = h * D_;
  const int half = lane >> 4, n = lane & 15;

  // Persistent Q fragments: 16 rows x 64 d  ->  two 16x32 A-frags
  const bf16_t* qptr = Q + rowbase + (size_t)(q0 + w * 16) * E_ + col0;
  const v16bf qf0 = frag_a(qptr, E_);
  const v16bf qf1 = frag_a(qptr + 32, E_);

  v8f o[4] = {};
  float mrow[8], lrow[8];
  #pragma unroll
  for (int r = 0; r < 8; ++r) { mrow[r] = -1e30f; lrow[r] = 0.f; }

  const int ntiles = (q0 + 128) >> 6;    // causal: keys <= q0+127, 64-key tiles
  for (int kt = 0; kt < ntiles; ++kt) {
    const int k0 = kt << 6;
    // stage K tile [64 keys][64 d] row-major: async global->LDS
    #pragma unroll
    for (int i = tid; i < 64 * 8; i += 256) {
      int j = i >> 3, dc = (i & 7) << 3;
      async_copy_b128(&Ks[j][dc],
                      &Kt[rowbase + (size_t)(k0 + j) * E_ + col0 + dc]);
    }
    // stage V tile transposed: Vs[d][key]
    for (int i = tid; i < 64 * 8; i += 256) {
      int j = i >> 3, dc = (i & 7) << 3;
      v8bf t = *(const v8bf*)&V[rowbase + (size_t)(k0 + j) * E_ + col0 + dc];
      #pragma unroll
      for (int e = 0; e < 8; ++e) Vs[dc + e][j] = t[e];
    }
    async_wait0();
    __syncthreads();

    // scores: S = Q(16x64) * K^T -> four 16x16 tiles, K-dim 64 = 2 chained WMMA
    v8f s[4];
    #pragma unroll
    for (int t = 0; t < 4; ++t) {
      v16bf b0 = frag_b(&Ks[t * 16][0],  72);
      v16bf b1 = frag_b(&Ks[t * 16][32], 72);
      v8f a = {};
      a = wmma_bf16(qf0, b0, a);
      a = wmma_bf16(qf1, b1, a);
      s[t] = a;
    }

    // online softmax over the 64-key tile (fp32 stats, wave32 shfl reductions)
    #pragma unroll
    for (int r = 0; r < 8; ++r) {
      const int qi = q0 + w * 16 + half * 8 + r;
      float v[4], vm = -1e30f;
      #pragma unroll
      for (int t = 0; t < 4; ++t) {
        v[t] = s[t][r] * 0.125f;                  // 1/sqrt(64)
        if (k0 + t * 16 + n > qi) v[t] = -1e30f;  // causal mask
        vm = fmaxf(vm, v[t]);
      }
      #pragma unroll
      for (int off = 1; off < 16; off <<= 1)
        vm = fmaxf(vm, __shfl_xor(vm, off, 32)); // row-max over 16 lanes of the half
      const float nm = fmaxf(mrow[r], vm);
      const float fr = __expf(mrow[r] - nm);
      float p[4], ps = 0.f;
      #pragma unroll
      for (int t = 0; t < 4; ++t) { p[t] = __expf(v[t] - nm); ps += p[t]; }
      #pragma unroll
      for (int off = 1; off < 16; off <<= 1)
        ps += __shfl_xor(ps, off, 32);           // row-sum
      lrow[r] = lrow[r] * fr + ps;
      mrow[r] = nm;
      #pragma unroll
      for (int dt = 0; dt < 4; ++dt) o[dt][r] *= fr;   // rescale running ctx
      #pragma unroll
      for (int t = 0; t < 4; ++t)
        Ps[w][half * 8 + r][t * 16 + n] = f2bf(p[t]);
    }

    // order P stores before frag reload (wave-private LDS region)
    asm volatile("s_wait_dscnt 0" ::: "memory");

    // ctx += P(16x64) * V(64x64): two A-frags, 2 chained WMMA per d-tile
    const v16bf pf0 = frag_a(&Ps[w][0][0],  72);
    const v16bf pf1 = frag_a(&Ps[w][0][32], 72);
    #pragma unroll
    for (int dt = 0; dt < 4; ++dt) {
      v16bf vb0 = frag_b(&Vs[dt * 16][0],  72);
      v16bf vb1 = frag_b(&Vs[dt * 16][32], 72);
      o[dt] = wmma_bf16(pf0, vb0, o[dt]);
      o[dt] = wmma_bf16(pf1, vb1, o[dt]);
    }
    __syncthreads();
  }

  // normalize and write ctx (bf16)
  #pragma unroll
  for (int r = 0; r < 8; ++r) {
    const float inv = 1.0f / lrow[r];
    const size_t row = rowbase + (size_t)(q0 + w * 16 + half * 8 + r) * E_ + col0;
    #pragma unroll
    for (int dt = 0; dt < 4; ++dt)
      ctx[row + dt * 16 + n] = f2bf(o[dt][r] * inv);
  }
}

// ---------------------------------------------------------------------------
// Host launch
// ---------------------------------------------------------------------------
extern "C" void kernel_launch(void* const* d_in, const int* in_sizes, int n_in,
                              void* d_out, int out_size, void* d_ws, size_t ws_size,
                              hipStream_t stream) {
  (void)in_sizes; (void)n_in; (void)out_size; (void)ws_size;
  const float* x  = (const float*)d_in[0];
  const float* qp = (const float*)d_in[1];
  const float* kp = (const float*)d_in[2];
  const float* vp = (const float*)d_in[3];
  const float* op = (const float*)d_in[4];
  float* out = (float*)d_out;

  char* ws = (char*)d_ws;
  bf16_t* xh = (bf16_t*)ws;                       // 4M elems (8 MB)
  bf16_t* wq = (bf16_t*)(ws + (size_t)(8u << 20));
  bf16_t* wk = wq + (1u << 20);
  bf16_t* wv = wk + (1u << 20);
  bf16_t* wo = wv + (1u << 20);
  bf16_t* Qb = (bf16_t*)(ws + (size_t)(16u << 20));
  bf16_t* Kb = Qb + (4u << 20);
  bf16_t* Vb = Kb + (4u << 20);
  bf16_t* Cb = Vb + (4u << 20);                   // total 48 MB

  const int nX = N_ * E_;       // 4096*1024
  const int nW = E_ * E_;       // 1024*1024
  cvt_f32_bf16<<<(nX + 255) / 256, 256, 0, stream>>>(x,  xh, nX);
  cvt_f32_bf16<<<(nW + 255) / 256, 256, 0, stream>>>(qp, wq, nW);
  cvt_f32_bf16<<<(nW + 255) / 256, 256, 0, stream>>>(kp, wk, nW);
  cvt_f32_bf16<<<(nW + 255) / 256, 256, 0, stream>>>(vp, wv, nW);
  cvt_f32_bf16<<<(nW + 255) / 256, 256, 0, stream>>>(op, wo, nW);

  dim3 ggrid(E_ / 128, N_ / 128);                 // (8, 32)
  gemm_bf16_kernel<bf16_t><<<ggrid, 256, 0, stream>>>(xh, wq, Qb, N_, E_, E_);
  gemm_bf16_kernel<bf16_t><<<ggrid, 256, 0, stream>>>(xh, wk, Kb, N_, E_, E_);
  gemm_bf16_kernel<bf16_t><<<ggrid, 256, 0, stream>>>(xh, wv, Vb, N_, E_, E_);

  attn_kernel<<<dim3(S_ / 128, H_, B_), 256, 0, stream>>>(Qb, Kb, Vb, Cb);

  gemm_bf16_kernel<float><<<ggrid, 256, 0, stream>>>(Cb, wo, out, N_, E_, E_);
}
